// MAM_29523605192767
// MI455X (gfx1250) — matrix operand order repaired and checked by reading, not compiled, not run on hardware
//
#include <hip/hip_runtime.h>
#include <stdint.h>

// ---------------------------------------------------------------------------
// Types & WMMA helpers (gfx1250, wave32)
// ---------------------------------------------------------------------------
typedef __bf16 bf16_t;
typedef __attribute__((ext_vector_type(16))) __bf16 v16bf;
typedef __attribute__((ext_vector_type(8)))  __bf16 v8bf;
typedef __attribute__((ext_vector_type(8)))  float  v8f;

struct v8bf_pair { v8bf lo, hi; };

static __device__ __forceinline__ v16bf join16(v8bf lo, v8bf hi) {
  v8bf_pair p{lo, hi};
  return __builtin_bit_cast(v16bf, p);
}

static __device__ __forceinline__ v8f zero8() {
  v8f z;
#pragma unroll
  for (int i = 0; i < 8; ++i) z[i] = 0.f;
  return z;
}

static __device__ __forceinline__ bf16_t f2bf(float f) {
  unsigned u = __builtin_bit_cast(unsigned, f);
  unsigned r = u + 0x7FFFu + ((u >> 16) & 1u);   // round-to-nearest-even
  unsigned short h = (unsigned short)(r >> 16);
  return __builtin_bit_cast(bf16_t, h);
}
static __device__ __forceinline__ float bf2f(bf16_t h) {
  unsigned short s = __builtin_bit_cast(unsigned short, h);
  unsigned u = ((unsigned)s) << 16;
  return __builtin_bit_cast(float, u);
}

// order-preserving float->uint key for atomicMin
static __device__ __forceinline__ unsigned fkey(float f) {
  unsigned u = __builtin_bit_cast(unsigned, f);
  return (u & 0x80000000u) ? ~u : (u | 0x80000000u);
}
static __device__ __forceinline__ float funkey(unsigned k) {
  unsigned u = (k & 0x80000000u) ? (k & 0x7FFFFFFFu) : ~k;
  return __builtin_bit_cast(float, u);
}

// A operand (16x32 bf16): lane<16 -> row=lane, K {0..7,16..23};
//                         lane>=16 -> row=lane-16, K {8..15,24..31}
// rowbase points at K=0 of this lane's row (K contiguous in memory).
static __device__ __forceinline__ v16bf frag_a(const bf16_t* rowbase, int lane) {
  int off = (lane < 16) ? 0 : 8;
  v8bf lo = *(const v8bf*)(rowbase + off);
  v8bf hi = *(const v8bf*)(rowbase + 16 + off);
  return join16(lo, hi);
}
// B operand (32x16 bf16): lane<16 -> col=lane, K {0..7,16..23}; lane>=16 -> col=lane-16, K {8..15,24..31}
// colbase points at K=0 of this lane's column (K contiguous in memory).
static __device__ __forceinline__ v16bf frag_b(const bf16_t* colbase, int lane) {
  const bf16_t* p = colbase + ((lane < 16) ? 0 : 8);
  v8bf lo = *(const v8bf*)(p);
  v8bf hi = *(const v8bf*)(p + 16);
  return join16(lo, hi);
}

static __device__ __forceinline__ v8f wmma_bf16(v16bf a, v16bf b, v8f c) {
  return __builtin_amdgcn_wmma_f32_16x16x32_bf16(false, a, false, b, (short)0, c,
                                                 false, false);
}

// ---------------------------------------------------------------------------
// Async global->LDS copy (CDNA5 Tensor/async path, tracked by ASYNCcnt).
// LDS destination address = low 32 bits of the generic pointer (flat-aperture
// mapping).
// ---------------------------------------------------------------------------
static __device__ __forceinline__ void async_g2l_b128(unsigned lds_off,
                                                      const void* gptr) {
  unsigned long long ga = (unsigned long long)(uintptr_t)gptr;
  asm volatile("global_load_async_to_lds_b128 %0, %1, off"
               :: "v"(lds_off), "v"(ga) : "memory");
}
#define WAIT_ASYNC_8() asm volatile("s_wait_asynccnt 0x8" ::: "memory")
#define WAIT_ASYNC_0() asm volatile("s_wait_asynccnt 0x0" ::: "memory")

// ---------------------------------------------------------------------------
// Prep kernels
// ---------------------------------------------------------------------------
__global__ void k_init(unsigned* mk, float* z) {
  int t = threadIdx.x;
  if (t < 16) { mk[t] = 0xFFFFFFFFu; z[t] = 0.f; }
}

__global__ void k_cvt(const float* __restrict__ src, bf16_t* __restrict__ dst, int n) {
  int i = blockIdx.x * 256 + threadIdx.x;
  if (i < n) dst[i] = f2bf(src[i]);
}

// x [b][64][1024] f32 -> xt [b][1024][64] bf16
__global__ void k_xT(const float* __restrict__ x, bf16_t* __restrict__ xt) {
  int i = blockIdx.x * 256 + threadIdx.x;
  if (i >= 16 * 1024 * 64) return;
  int c = i & 63, p = (i >> 6) & 1023, b = i >> 16;
  xt[i] = f2bf(x[((size_t)b * 64 + c) * 1024 + p]);
}

// ---------------------------------------------------------------------------
// Projection GEMM: out = W(64x64) @ x + bias. Xt: [b][1024][64] bf16.
// mode 0: out[b][o][p] (for K/V side)   mode 1: out[b][p][o] (for Q side)
// One wave computes a 16(o) x 64(p) strip: 4 accumulators, A reused.
// ---------------------------------------------------------------------------
__global__ void k_proj(const bf16_t* __restrict__ Wb, const bf16_t* __restrict__ Xt,
                       const float* __restrict__ bias, bf16_t* __restrict__ out,
                       int mode) {
  const int I = 64;
  int lane = threadIdx.x;
  int b = blockIdx.x, pb = blockIdx.y, ob = blockIdx.z;
  int o0 = ob * 16, p0 = pb * 64;
  int nn = lane & 15;
  v8f acc[4] = {zero8(), zero8(), zero8(), zero8()};
  int row = o0 + nn;
#pragma unroll
  for (int kc = 0; kc < I; kc += 32) {
    v16bf aA = frag_a(Wb + (size_t)row * I + kc, lane);
#pragma unroll
    for (int t = 0; t < 4; ++t) {
      const bf16_t* colbase = Xt + ((size_t)b * 1024 + p0 + t * 16 + nn) * I + kc;
      acc[t] = wmma_bf16(aA, frag_b(colbase, lane), acc[t]);
    }
  }
  int mo = (lane < 16) ? 0 : 8;
#pragma unroll
  for (int t = 0; t < 4; ++t)
#pragma unroll
    for (int r = 0; r < 8; ++r) {
      int o = o0 + r + mo;
      int p = p0 + t * 16 + nn;
      float v = acc[t][r] + bias[o];
      size_t idx = mode ? ((size_t)b * 1024 + p) * 64 + o
                        : ((size_t)b * 64 + o) * 1024 + p;
      out[idx] = f2bf(v);
    }
}

// ---------------------------------------------------------------------------
// Generic block GEMM: Y[b][O][1024] f32 = W(OxI) @ act, act [b][1024][I] bf16
// The activation strip (64 p-columns x 32 K) is staged global->LDS with
// double-buffered async DMA; ASYNCcnt pipelining hides global latency behind
// the WMMAs. Single wave32 per workgroup -> no barriers needed (async writes
// to a buffer are only issued after its previous readers' ds_loads drained).
// ---------------------------------------------------------------------------
__global__ void k_gemm(const bf16_t* __restrict__ Wb, const bf16_t* __restrict__ Xt,
                       float* __restrict__ Y, int O, int I) {
  __shared__ __align__(16) bf16_t Xs[2][64 * 32];   // 2 x 4KB tiles
  int lane = threadIdx.x;
  int b = blockIdx.x, pb = blockIdx.y, ob = blockIdx.z;
  int o0 = ob * 16, p0 = pb * 64;
  int nn = lane & 15;
  v8f acc[4] = {zero8(), zero8(), zero8(), zero8()};
  int row = o0 + nn;

  const bf16_t* xbase = Xt + ((size_t)b * 1024 + p0) * I;   // 64-row strip
  unsigned seg = (unsigned)lane;          // this lane's 16B segment index base
  unsigned r0 = seg >> 2;                 // row of first segment (4 segs/row)
  unsigned e0 = (seg & 3) * 8;            // element offset within 64B row

  // stage one 64x32 K-chunk into Xs[buf]: 256 16B segments, 8 issues/lane
  auto stage = [&](int buf, int kc) {
    unsigned dst0 = (unsigned)(uintptr_t)(void*)&Xs[buf][0];
#pragma unroll
    for (int it = 0; it < 8; ++it) {
      unsigned r = r0 + it * 8;           // +32 segments == +8 rows
      const bf16_t* g = xbase + (size_t)r * I + kc + e0;
      async_g2l_b128(dst0 + r * 64u + e0 * 2u, g);
    }
  };

  int nch = I / 32;
  stage(0, 0);
  for (int ic = 0; ic < nch; ++ic) {
    int kc = ic * 32;
    if (ic + 1 < nch) {
      stage((ic + 1) & 1, kc + 32);       // prefetch next chunk
      WAIT_ASYNC_8();                     // oldest 8 (current chunk) landed
    } else {
      WAIT_ASYNC_0();
    }
    v16bf aA = frag_a(Wb + (size_t)row * I + kc, lane);
    const bf16_t* xs = &Xs[ic & 1][0];
#pragma unroll
    for (int t = 0; t < 4; ++t)
      acc[t] = wmma_bf16(aA, frag_b(xs + (t * 16 + nn) * 32, lane), acc[t]);
  }

  int mo = (lane < 16) ? 0 : 8;
#pragma unroll
  for (int t = 0; t < 4; ++t)
#pragma unroll
    for (int r = 0; r < 8; ++r)
      Y[((size_t)b * O + o0 + r + mo) * 1024 + p0 + t * 16 + nn] = acc[t][r];
}

// ---------------------------------------------------------------------------
// Energy min pass: e = qa^T qb tiles, global per-batch min via keyed atomicMin.
// Qt/Kt: [b][1024][64] bf16. One wave owns a 16-row i-block, loops all j.
// ---------------------------------------------------------------------------
__global__ void k_emin(const bf16_t* __restrict__ Qt, const bf16_t* __restrict__ Kt,
                       unsigned* __restrict__ minkey) {
  int lane = threadIdx.x;
  int b = blockIdx.x, ib = blockIdx.y;
  int i0 = ib * 16, nn = lane & 15;
  const bf16_t* qrow = Qt + ((size_t)b * 1024 + i0 + nn) * 64;
  v16bf aQ0 = frag_a(qrow, lane);
  v16bf aQ1 = frag_a(qrow + 32, lane);
  float lmin = 3.4e38f;
  for (int j0 = 0; j0 < 1024; j0 += 16) {
    const bf16_t* kcol = Kt + ((size_t)b * 1024 + j0 + nn) * 64;
    v8f e = wmma_bf16(aQ0, frag_b(kcol, lane), zero8());
    e = wmma_bf16(aQ1, frag_b(kcol + 32, lane), e);
#pragma unroll
    for (int r = 0; r < 8; ++r) lmin = fminf(lmin, e[r]);
  }
  atomicMin(minkey + b, fkey(lmin));
}

// ---------------------------------------------------------------------------
// Fused attention pass (flash-style, energy recomputed, never materialized):
//   p_ij = exp(min_b - e_ij),  U[i][c] += sum_j p_ij * V[c][j],  Z_b += sum p
// Qt,Kt: [b][1024][64] bf16 (row-major "T"), Vc: [b][64][1024] bf16.
// U out: [b][1024][64] f32 (unnormalized; epilogue divides by Z).
// P tile transposed D-layout -> A-layout via 1KB LDS (single wave32 workgroup).
// Second launch with (Q<->K, V=ka) computes the transposed application.
// ---------------------------------------------------------------------------
__global__ void k_attn(const bf16_t* __restrict__ Qt, const bf16_t* __restrict__ Kt,
                       const bf16_t* __restrict__ Vc, float* __restrict__ U,
                       float* __restrict__ Zbuf, const unsigned* __restrict__ minkey,
                       int zflag) {
  __shared__ __align__(16) bf16_t P[16 * 32];
  int lane = threadIdx.x;
  int b = blockIdx.x, ib = blockIdx.y;
  int i0 = ib * 16, nn = lane & 15;
  int mo = (lane < 16) ? 0 : 8;
  float mval = funkey(minkey[b]);
  const bf16_t* qrow = Qt + ((size_t)b * 1024 + i0 + nn) * 64;
  v16bf aQ0 = frag_a(qrow, lane);
  v16bf aQ1 = frag_a(qrow + 32, lane);
  v8f acc[4] = {zero8(), zero8(), zero8(), zero8()};
  float zpart = 0.f;
  for (int j0 = 0; j0 < 1024; j0 += 32) {
    const bf16_t* kc0 = Kt + ((size_t)b * 1024 + j0 + nn) * 64;
    const bf16_t* kc1 = Kt + ((size_t)b * 1024 + j0 + 16 + nn) * 64;
    v8f e0 = wmma_bf16(aQ0, frag_b(kc0, lane), zero8());
    e0 = wmma_bf16(aQ1, frag_b(kc0 + 32, lane), e0);
    v8f e1 = wmma_bf16(aQ0, frag_b(kc1, lane), zero8());
    e1 = wmma_bf16(aQ1, frag_b(kc1 + 32, lane), e1);
    __syncthreads();                       // WAR vs previous iter's LDS reads
#pragma unroll
    for (int r = 0; r < 8; ++r) {
      float p0 = __expf(mval - e0[r]);
      float p1 = __expf(mval - e1[r]);
      zpart += p0 + p1;
      P[(r + mo) * 32 + nn]      = f2bf(p0);
      P[(r + mo) * 32 + 16 + nn] = f2bf(p1);
    }
    __syncthreads();                       // RAW: make P visible before A-load
    v16bf aP = frag_a(&P[nn * 32], lane);  // P block as 16x32 A operand
#pragma unroll
    for (int ci = 0; ci < 4; ++ci) {
      const bf16_t* vcol = Vc + ((size_t)b * 64 + ci * 16 + nn) * 1024 + j0;
      acc[ci] = wmma_bf16(aP, frag_b(vcol, lane), acc[ci]);
    }
  }
  if (zflag) atomicAdd(Zbuf + b, zpart);
#pragma unroll
  for (int ci = 0; ci < 4; ++ci)
#pragma unroll
    for (int r = 0; r < 8; ++r)
      U[((size_t)b * 1024 + i0 + r + mo) * 64 + ci * 16 + nn] = acc[ci][r];
}

// ---------------------------------------------------------------------------
// BatchNorm stats: per-channel mean/var over (B,N)=16384 -> alpha/beta
// ---------------------------------------------------------------------------
__global__ void k_stats(const float* __restrict__ T, const float* __restrict__ s,
                        const float* __restrict__ bb, float* __restrict__ alpha,
                        float* __restrict__ beta, int O) {
  __shared__ float red[512];
  int c = blockIdx.x, t = threadIdx.x;
  float sum = 0.f, sq = 0.f;
  for (int m = t; m < 16 * 1024; m += 256) {
    int b = m >> 10, p = m & 1023;
    float v = T[((size_t)b * O + c) * 1024 + p];
    sum += v; sq += v * v;
  }
  red[t] = sum; red[256 + t] = sq;
  __syncthreads();
  for (int st = 128; st > 0; st >>= 1) {
    if (t < st) { red[t] += red[t + st]; red[256 + t] += red[256 + t + st]; }
    __syncthreads();
  }
  if (t == 0) {
    float mean = red[0] * (1.f / 16384.f);
    float var  = red[256] * (1.f / 16384.f) - mean * mean;
    float a = s[c] * rsqrtf(var + 1e-5f);
    alpha[c] = a;
    beta[c]  = bb[c] - mean * a;
  }
}

// ---------------------------------------------------------------------------
// Epilogue / elementwise kernels
// ---------------------------------------------------------------------------
// ya0 = concat(x, U/Z) -> act [b][1024][128] bf16
__global__ void k_act0(const float* __restrict__ x, const float* __restrict__ U,
                       const float* __restrict__ Z, bf16_t* __restrict__ act) {
  int i = blockIdx.x * 256 + threadIdx.x;
  if (i >= 16 * 1024 * 128) return;
  int c = i & 127, p = (i >> 7) & 1023, b = i >> 17;
  float v = (c < 64) ? x[((size_t)b * 64 + c) * 1024 + p]
                     : U[((size_t)b * 1024 + p) * 64 + (c - 64)] / Z[b];
  act[i] = f2bf(v);
}

// act = relu(alpha*T + beta), T [b][O][1024] f32 -> act [b][1024][O] bf16
__global__ void k_bnrelu(const float* __restrict__ T, const float* __restrict__ alpha,
                         const float* __restrict__ beta, bf16_t* __restrict__ act,
                         int O) {
  int i = blockIdx.x * 256 + threadIdx.x;
  if (i >= 16 * 1024 * O) return;
  int c = i % O, p = (i / O) & 1023, b = i / (O * 1024);
  float v = alpha[c] * T[((size_t)b * O + c) * 1024 + p] + beta[c];
  act[i] = f2bf(fmaxf(v, 0.f));
}

// out = relu(bn(T2) + shortcut); shortcut is bn(Ts) if Ts, else actin (identity)
__global__ void k_bnaddrelu(const float* __restrict__ T2, const float* __restrict__ a2,
                            const float* __restrict__ b2, const float* __restrict__ Ts,
                            const float* __restrict__ as_, const float* __restrict__ bs_,
                            const bf16_t* __restrict__ actin, bf16_t* __restrict__ actout,
                            int O) {
  int i = blockIdx.x * 256 + threadIdx.x;
  if (i >= 16 * 1024 * O) return;
  int c = i % O, p = (i / O) & 1023, b = i / (O * 1024);
  size_t ti = ((size_t)b * O + c) * 1024 + p;
  float sc = Ts ? (as_[c] * Ts[ti] + bs_[c]) : bf2f(actin[i]);
  float v = a2[c] * T2[ti] + b2[c] + sc;
  actout[i] = f2bf(fmaxf(v, 0.f));
}

// final head: out[b][c][p] = alpha*T + beta  (layouts identical, linear copy)
__global__ void k_head(const float* __restrict__ T, const float* __restrict__ alpha,
                       const float* __restrict__ beta, float* __restrict__ out) {
  int i = blockIdx.x * 256 + threadIdx.x;
  if (i >= 16 * 64 * 1024) return;
  int c = (i >> 10) & 63;
  out[i] = alpha[c] * T[i] + beta[c];
}

// ---------------------------------------------------------------------------
// Host launcher
// ---------------------------------------------------------------------------
extern "C" void kernel_launch(void* const* d_in, const int* in_sizes, int n_in,
                              void* d_out, int out_size, void* d_ws, size_t ws_size,
                              hipStream_t stream) {
  (void)in_sizes; (void)out_size; (void)ws_size;
  const float* F[46];
  for (int i = 0; i < 46 && i < n_in; ++i) F[i] = (const float*)d_in[i];

  char* base = (char*)d_ws;
  size_t cur = 0;
  auto alloc = [&](size_t bytes) -> void* {
    void* p = base + cur;
    cur = (cur + bytes + 255) & ~(size_t)255;
    return p;
  };

  // bf16 weight copies: wqa,wqb,wka,wkb, a1w1,a1w2, b1w1,b1w2,
  //                     a2w1,a2w2,a2ws, b2w1,b2w2,b2ws, oaw,obw
  static const int widx[16] = {2, 4, 6, 8, 10, 13, 16, 19, 22, 25, 28, 31, 34, 37, 40, 43};
  static const int wcnt[16] = {4096, 4096, 4096, 4096, 16384, 16384, 16384, 16384,
                               32768, 65536, 32768, 32768, 65536, 32768, 16384, 16384};
  bf16_t* WB[16];
  for (int i = 0; i < 16; ++i) WB[i] = (bf16_t*)alloc((size_t)wcnt[i] * 2);

  const size_t NB64 = (size_t)16 * 1024 * 64;
  bf16_t* xaT = (bf16_t*)alloc(NB64 * 2);
  bf16_t* xbT = (bf16_t*)alloc(NB64 * 2);
  bf16_t* qaT = (bf16_t*)alloc(NB64 * 2);
  bf16_t* qbT = (bf16_t*)alloc(NB64 * 2);
  bf16_t* kaC = (bf16_t*)alloc(NB64 * 2);   // [b][64][1024]
  bf16_t* kbC = (bf16_t*)alloc(NB64 * 2);
  float*  UaT = (float*)alloc(NB64 * 4);    // [b][1024][64]
  float*  UbT = (float*)alloc(NB64 * 4);
  unsigned* minkey = (unsigned*)alloc(64);
  float*  Z   = (float*)alloc(64);
  float*  ab  = (float*)alloc(2048 * 4);    // alpha/beta slots
  float*  t1  = (float*)alloc((size_t)16 * 256 * 1024 * 4);
  float*  t2  = (float*)alloc((size_t)16 * 256 * 1024 * 4);
  float*  ts  = (float*)alloc((size_t)16 * 256 * 1024 * 4);
  bf16_t* act0 = (bf16_t*)alloc((size_t)16 * 1024 * 256 * 2);
  bf16_t* act1 = (bf16_t*)alloc((size_t)16 * 1024 * 256 * 2);
  bf16_t* act2 = (bf16_t*)alloc((size_t)16 * 1024 * 256 * 2);

  dim3 w32(32, 1, 1);

  k_init<<<1, 32, 0, stream>>>(minkey, Z);
  for (int i = 0; i < 16; ++i)
    k_cvt<<<(wcnt[i] + 255) / 256, 256, 0, stream>>>(F[widx[i]], WB[i], wcnt[i]);

  int nx = 16 * 1024 * 64;
  k_xT<<<(nx + 255) / 256, 256, 0, stream>>>(F[0], xaT);
  k_xT<<<(nx + 255) / 256, 256, 0, stream>>>(F[1], xbT);

  dim3 gp(16, 16, 4);
  k_proj<<<gp, w32, 0, stream>>>(WB[0], xaT, F[3], qaT, 1);  // qa -> [n][c]
  k_proj<<<gp, w32, 0, stream>>>(WB[1], xbT, F[5], qbT, 1);  // qb -> [n][c]
  k_proj<<<gp, w32, 0, stream>>>(WB[2], xaT, F[7], kaC, 0);  // ka -> [c][n]
  k_proj<<<gp, w32, 0, stream>>>(WB[3], xbT, F[9], kbC, 0);  // kb -> [c][n]

  dim3 ga(16, 64, 1);
  k_emin<<<ga, w32, 0, stream>>>(qaT, qbT, minkey);
  k_attn<<<ga, w32, 0, stream>>>(qaT, qbT, kbC, UaT, Z, minkey, 1);  // + Z
  k_attn<<<ga, w32, 0, stream>>>(qbT, qaT, kaC, UbT, Z, minkey, 0);  // e^T side

  auto run_block = [&](const bf16_t* w1, const float* s1, const float* b1p,
                       const bf16_t* w2, const float* s2, const float* b2p,
                       const bf16_t* wsb, const float* ssb, const float* bsb,
                       const bf16_t* ain, bf16_t* ahid, bf16_t* aout,
                       int cin, int cout) {
    dim3 g(16, 16, cout / 16);
    int tot = 16 * 1024 * cout;
    k_gemm<<<g, w32, 0, stream>>>(w1, ain, t1, cout, cin);
    k_stats<<<cout, 256, 0, stream>>>(t1, s1, b1p, ab + 0, ab + 256, cout);
    k_bnrelu<<<(tot + 255) / 256, 256, 0, stream>>>(t1, ab + 0, ab + 256, ahid, cout);
    k_gemm<<<g, w32, 0, stream>>>(w2, ahid, t2, cout, cout);
    k_stats<<<cout, 256, 0, stream>>>(t2, s2, b2p, ab + 512, ab + 768, cout);
    const float* tsp = nullptr;
    if (wsb) {
      k_gemm<<<g, w32, 0, stream>>>(wsb, ain, ts, cout, cin);
      k_stats<<<cout, 256, 0, stream>>>(ts, ssb, bsb, ab + 1024, ab + 1280, cout);
      tsp = ts;
    }
    k_bnaddrelu<<<(tot + 255) / 256, 256, 0, stream>>>(
        t2, ab + 512, ab + 768, tsp, ab + 1024, ab + 1280,
        wsb ? nullptr : ain, aout, cout);
  };

  auto run_head = [&](const bf16_t* wo, const float* so, const float* bo,
                      const bf16_t* ain, float* outp) {
    dim3 g(16, 16, 4);
    k_gemm<<<g, w32, 0, stream>>>(wo, ain, t1, 64, 256);
    k_stats<<<64, 256, 0, stream>>>(t1, so, bo, ab + 0, ab + 256, 64);
    k_head<<<(16 * 64 * 1024 + 255) / 256, 256, 0, stream>>>(t1, ab + 0, ab + 256, outp);
  };

  float* out = (float*)d_out;
  int nact0 = 16 * 1024 * 128;

  // ---- side A ----
  k_act0<<<(nact0 + 255) / 256, 256, 0, stream>>>(F[0], UaT, Z, act0);
  run_block(WB[4], F[11], F[12], WB[5], F[14], F[15],
            nullptr, nullptr, nullptr, act0, act1, act2, 128, 128);
  run_block(WB[8], F[23], F[24], WB[9], F[26], F[27],
            WB[10], F[29], F[30], act2, act1, act0, 128, 256);
  run_head(WB[14], F[41], F[42], act0, out);

  // ---- side B ----
  k_act0<<<(nact0 + 255) / 256, 256, 0, stream>>>(F[1], UbT, Z, act0);
  run_block(WB[6], F[17], F[18], WB[7], F[20], F[21],
            nullptr, nullptr, nullptr, act0, act1, act2, 128, 128);
  run_block(WB[11], F[32], F[33], WB[12], F[35], F[36],
            WB[13], F[38], F[39], act2, act1, act0, 128, 256);
  run_head(WB[15], F[44], F[45], act0, out + 16 * 64 * 1024);
}
